// SNN_39470749450737
// MI455X (gfx1250) — matrix operand order repaired and checked by reading, compiled, not run on hardware
//
#include <hip/hip_runtime.h>

// ---------------------------------------------------------------------------
// EP/SNN free-phase relaxation, 30 Euler steps, batch 256.
// Heavy ops (256x1024x1024 GEMMs, and the hoisted 256x1024x784 GEMM) run on
// v_wmma_f32_16x16x32_f16 with register double-buffering so fragment loads
// for tile k+32 overlap the WMMAs of tile k. States/accumulators stay fp32.
// ---------------------------------------------------------------------------

typedef __attribute__((ext_vector_type(16))) _Float16 v16h;
typedef __attribute__((ext_vector_type(8)))  _Float16 v8h;
typedef __attribute__((ext_vector_type(8)))  float    v8f;

#define BATCH   256
#define L0      10
#define L1      1024
#define L2N     1024
#define D3      784
#define KP      800          // 784 padded up to a multiple of 32
#define NSTEPS  30
#define DT_C    0.2f

#define CNT0    (BATCH * L0)            // 2560
#define CNT1    (BATCH * L1)            // 262144
#define CNT2    (BATCH * L2N)           // 262144
#define CNT01   (CNT0 + CNT1)           // 264704
#define CNT012  (CNT0 + CNT1 + CNT2)    // 526848
#define CNTDATA (BATCH * D3)            // 200704
#define CNTOUT  (CNT012 + CNTDATA)      // 727552

__device__ __forceinline__ unsigned hashu(unsigned x) {
    x ^= x >> 16; x *= 0x7feb352du;
    x ^= x >> 15; x *= 0x846ca68bu;
    x ^= x >> 16; return x;
}

__device__ __forceinline__ float clip01(float x) {
    return fminf(fmaxf(x, 0.0f), 1.0f);
}

// A fragment (16x32, MxK): lanes 0-15 hold row M=lr with k in {0..15},
// lanes 16-31 hold the same rows with k in {16..31}.
__device__ __forceinline__ v16h load_fragA(const _Float16* __restrict__ arow, int k) {
    v8h alo = *(const v8h*)(arow + k);
    v8h ahi = *(const v8h*)(arow + k + 8);
    v16h a;
#pragma unroll
    for (int i = 0; i < 8; ++i) { a[i] = alo[i]; a[i + 8] = ahi[i]; }
    return a;
}

// ---------------------------------------------------------------------------
// Copy initial fp32 states from inputs into workspace (inputs are immutable).
// ---------------------------------------------------------------------------
__global__ void init_states(const float* __restrict__ s0, const float* __restrict__ s1,
                            const float* __restrict__ s2,
                            float* __restrict__ s0f, float* __restrict__ s1f,
                            float* __restrict__ s2f) {
    int i = blockIdx.x * blockDim.x + threadIdx.x;
    if (i < CNT0)        s0f[i]          = s0[i];
    else if (i < CNT01)  s1f[i - CNT0]   = s1[i - CNT0];
    else if (i < CNT012) s2f[i - CNT01]  = s2[i - CNT01];
}

// ---------------------------------------------------------------------------
// fp32 -> fp16 conversion with optional K padding (zeros) for WMMA operands.
// ---------------------------------------------------------------------------
__global__ void cvt_pad(_Float16* __restrict__ dst, const float* __restrict__ src,
                        int rows, int scols, int dcols) {
    int i = blockIdx.x * blockDim.x + threadIdx.x;
    if (i >= rows * dcols) return;
    int r = i / dcols, c = i - r * dcols;
    dst[i] = (c < scols) ? (_Float16)src[r * scols + c] : (_Float16)0.0f;
}

// ---------------------------------------------------------------------------
// Bernoulli spikes from pre-update states + rho(s1) staging (all fp16).
// ---------------------------------------------------------------------------
__global__ void spikes_kernel(const float* __restrict__ s0f, const float* __restrict__ s1f,
                              const float* __restrict__ s2f,
                              _Float16* __restrict__ sp0, _Float16* __restrict__ sp1,
                              _Float16* __restrict__ sp2, _Float16* __restrict__ rho1,
                              int t) {
    int i = blockIdx.x * blockDim.x + threadIdx.x;
    if (i >= CNT012) return;
    int layer, elem; float s;
    if (i < CNT0)        { layer = 0; elem = i;          s = s0f[elem]; }
    else if (i < CNT01)  { layer = 1; elem = i - CNT0;   s = s1f[elem]; }
    else                 { layer = 2; elem = i - CNT01;  s = s2f[elem]; }
    float r = clip01(s);
    unsigned h = hashu((unsigned)elem * 0x9E3779B9u
                       ^ hashu((unsigned)t * 3u + (unsigned)layer + 0x68bc21ebu));
    float u = (float)(h >> 8) * (1.0f / 16777216.0f);
    _Float16 sp = (_Float16)((u < r) ? 1.0f : 0.0f);
    if (layer == 0)      sp0[elem] = sp;
    else if (layer == 1) { sp1[elem] = sp; rho1[elem] = (_Float16)r; }
    else                 sp2[elem] = sp;
}

// ---------------------------------------------------------------------------
// WMMA GEMM: C[m,n] = sum_k A[m,k] * B[n,k]   (i.e. C = Act @ W.T)
// One wave -> 16x64 output strip (1 A frag, 4 B frags, 4 f32 accumulators).
// 8 waves/CTA, 32 CTAs cover 256x1024. blockIdx.y selects operand pair so the
// two independent per-step GEMMs share one launch. Register double-buffering:
// fragments for k+32 are loaded before the WMMAs for k issue, so waits are
// partial instead of full drains. Uniform control flow (EXEC all ones).
// ---------------------------------------------------------------------------
__global__ void gemm_wmma_dual(const _Float16* __restrict__ Aa, const _Float16* __restrict__ Ba,
                               float* __restrict__ Ca,
                               const _Float16* __restrict__ Ab, const _Float16* __restrict__ Bb,
                               float* __restrict__ Cb,
                               int lda, int ldb, int ldc, int N, int K) {
    const _Float16* A; const _Float16* B; float* C;
    if (blockIdx.y == 0) { A = Aa; B = Ba; C = Ca; }
    else                 { A = Ab; B = Bb; C = Cb; }

    const int lane    = threadIdx.x & 31;
    const int wave    = threadIdx.x >> 5;
    const int strip   = blockIdx.x * 8 + wave;
    const int nStrips = N >> 6;                 // strips along N (64 wide)
    const int m0 = (strip / nStrips) << 4;
    const int n0 = (strip % nStrips) << 6;
    const int lr = lane & 15;
    const int hi = lane >> 4;

    v8f z = {};
    v8f c0 = z, c1 = z, c2 = z, c3 = z;

    const _Float16* arow = A + (size_t)(m0 + lr) * lda + hi * 16;
    // B fragment (32x16, KxN) from W rows: lane lr -> column N=lr,
    // 16 contiguous k starting at k + hi*16.
    const _Float16* b0p = B + (size_t)(n0 + lr) * ldb + hi * 16;
    const _Float16* b1p = b0p + (size_t)16 * ldb;
    const _Float16* b2p = b0p + (size_t)32 * ldb;
    const _Float16* b3p = b0p + (size_t)48 * ldb;

    // Prologue: fragments for k = 0.
    v16h a  = load_fragA(arow, 0);
    v16h b0 = *(const v16h*)(b0p);
    v16h b1 = *(const v16h*)(b1p);
    v16h b2 = *(const v16h*)(b2p);
    v16h b3 = *(const v16h*)(b3p);

    // Pipelined main loop: issue k-tile loads, compute (k-32)-tile WMMAs.
    for (int k = 32; k < K; k += 32) {
        v16h an  = load_fragA(arow, k);
        v16h bn0 = *(const v16h*)(b0p + k);
        v16h bn1 = *(const v16h*)(b1p + k);
        v16h bn2 = *(const v16h*)(b2p + k);
        v16h bn3 = *(const v16h*)(b3p + k);
        c0 = __builtin_amdgcn_wmma_f32_16x16x32_f16(false, a, false, b0, (short)0, c0, false, false);
        c1 = __builtin_amdgcn_wmma_f32_16x16x32_f16(false, a, false, b1, (short)0, c1, false, false);
        c2 = __builtin_amdgcn_wmma_f32_16x16x32_f16(false, a, false, b2, (short)0, c2, false, false);
        c3 = __builtin_amdgcn_wmma_f32_16x16x32_f16(false, a, false, b3, (short)0, c3, false, false);
        a = an; b0 = bn0; b1 = bn1; b2 = bn2; b3 = bn3;
    }
    // Epilogue: final tile.
    c0 = __builtin_amdgcn_wmma_f32_16x16x32_f16(false, a, false, b0, (short)0, c0, false, false);
    c1 = __builtin_amdgcn_wmma_f32_16x16x32_f16(false, a, false, b1, (short)0, c1, false, false);
    c2 = __builtin_amdgcn_wmma_f32_16x16x32_f16(false, a, false, b2, (short)0, c2, false, false);
    c3 = __builtin_amdgcn_wmma_f32_16x16x32_f16(false, a, false, b3, (short)0, c3, false, false);

    // C/D layout: VGPR r -> (M = hi*8 + r, N = lane&15)
    float* crow = C + (size_t)(m0 + hi * 8) * ldc + n0 + lr;
#pragma unroll
    for (int r = 0; r < 8; ++r) {
        crow[(size_t)r * ldc + 0]  = c0[r];
        crow[(size_t)r * ldc + 16] = c1[r];
        crow[(size_t)r * ldc + 32] = c2[r];
        crow[(size_t)r * ldc + 48] = c3[r];
    }
}

// ---------------------------------------------------------------------------
// Tiny GEMM: A0[b,n] = sum_k sp1[b,k] * W0[n,k]   (256x10, K=1024) — fp32.
// ---------------------------------------------------------------------------
__global__ void a0_gemm(const _Float16* __restrict__ sp1, const float* __restrict__ W0,
                        float* __restrict__ A0) {
    int i = blockIdx.x * blockDim.x + threadIdx.x;
    if (i >= CNT0) return;
    int b = i / L0, n = i - b * L0;
    const _Float16* x = sp1 + (size_t)b * L1;
    const float* w = W0 + (size_t)n * L1;
    float acc = 0.0f;
    for (int k = 0; k < L1; ++k) acc += (float)x[k] * w[k];
    A0[i] = acc;
}

// ---------------------------------------------------------------------------
// Euler update + clip. The K=10 term sp0 @ W1.T is folded in here (fp32).
// ---------------------------------------------------------------------------
__global__ void update_kernel(float* __restrict__ s0f, float* __restrict__ s1f,
                              float* __restrict__ s2f,
                              const float* __restrict__ A0, const float* __restrict__ A1,
                              const float* __restrict__ A2, const float* __restrict__ P2,
                              const float* __restrict__ b0, const float* __restrict__ b2,
                              const float* __restrict__ b4,
                              const _Float16* __restrict__ sp0, const float* __restrict__ W1) {
    int i = blockIdx.x * blockDim.x + threadIdx.x;
    if (i < CNT0) {
        int n = i % L0;
        float s = s0f[i];
        float ds = -s + A0[i] + b0[n];
        s0f[i] = clip01(s + DT_C * ds);
    } else if (i < CNT01) {
        int e = i - CNT0;
        int b = e / L1, n = e - b * L1;
        float s = s1f[e];
        float acc = A1[e] + b2[n];
        const _Float16* sp = sp0 + (size_t)b * L0;
        const float* w = W1 + (size_t)n * L0;
#pragma unroll
        for (int k = 0; k < L0; ++k) acc += (float)sp[k] * w[k];
        s1f[e] = clip01(s + DT_C * (-s + acc));
    } else if (i < CNT012) {
        int e = i - CNT01;
        int n = e % L2N;
        float s = s2f[e];
        float ds = -s + P2[e] + b4[n] + A2[e];
        s2f[e] = clip01(s + DT_C * ds);
    }
}

// ---------------------------------------------------------------------------
// Pack outputs: (s0, s1, s2, data) concatenated flat, fp32.
// ---------------------------------------------------------------------------
__global__ void final_copy(float* __restrict__ out,
                           const float* __restrict__ s0f, const float* __restrict__ s1f,
                           const float* __restrict__ s2f, const float* __restrict__ data) {
    int i = blockIdx.x * blockDim.x + threadIdx.x;
    if (i < CNT0)         out[i] = s0f[i];
    else if (i < CNT01)   out[i] = s1f[i - CNT0];
    else if (i < CNT012)  out[i] = s2f[i - CNT01];
    else if (i < CNTOUT)  out[i] = data[i - CNT012];
}

// ---------------------------------------------------------------------------
// Host launcher. All launches on `stream`; workspace carved by bump allocator.
// ---------------------------------------------------------------------------
extern "C" void kernel_launch(void* const* d_in, const int* in_sizes, int n_in,
                              void* d_out, int out_size, void* d_ws, size_t ws_size,
                              hipStream_t stream) {
    (void)in_sizes; (void)n_in; (void)out_size; (void)ws_size;

    const float* data = (const float*)d_in[0];
    const float* s0   = (const float*)d_in[1];
    const float* s1   = (const float*)d_in[2];
    const float* s2   = (const float*)d_in[3];
    const float* W0   = (const float*)d_in[4];
    const float* b0   = (const float*)d_in[5];
    const float* W1   = (const float*)d_in[6];
    const float* W2   = (const float*)d_in[7];
    const float* b2   = (const float*)d_in[8];
    const float* W3   = (const float*)d_in[9];
    const float* W4   = (const float*)d_in[10];
    const float* b4   = (const float*)d_in[11];

    char* w = (char*)d_ws;
    size_t off = 0;
    auto alloc = [&](size_t bytes) -> char* {
        char* p = w + off;
        off = (off + bytes + 255) & ~(size_t)255;
        return p;
    };

    float*    s0f   = (float*)alloc(CNT0 * 4);
    float*    s1f   = (float*)alloc(CNT1 * 4);
    float*    s2f   = (float*)alloc(CNT2 * 4);
    float*    A0    = (float*)alloc(CNT0 * 4);
    float*    A1    = (float*)alloc(CNT1 * 4);
    float*    A2    = (float*)alloc(CNT2 * 4);
    float*    P2    = (float*)alloc(CNT2 * 4);
    _Float16* sp0h  = (_Float16*)alloc(CNT0 * 2);
    _Float16* sp1h  = (_Float16*)alloc(CNT1 * 2);
    _Float16* sp2h  = (_Float16*)alloc(CNT2 * 2);
    _Float16* rho1h = (_Float16*)alloc(CNT1 * 2);
    _Float16* W2h   = (_Float16*)alloc((size_t)L1 * L2N * 2);
    _Float16* W3h   = (_Float16*)alloc((size_t)L2N * L1 * 2);
    _Float16* W4h   = (_Float16*)alloc((size_t)L2N * KP * 2);
    _Float16* datah = (_Float16*)alloc((size_t)BATCH * KP * 2);

    const int TPB = 256;
    const int gAll = (CNT012 + TPB - 1) / TPB;   // 2058

    // --- one-time setup (stream-ordered) ---
    init_states<<<gAll, TPB, 0, stream>>>(s0, s1, s2, s0f, s1f, s2f);
    cvt_pad<<<(L1 * L2N) / TPB, TPB, 0, stream>>>(W2h, W2, L1, L2N, L2N);
    cvt_pad<<<(L2N * L1) / TPB, TPB, 0, stream>>>(W3h, W3, L2N, L1, L1);
    cvt_pad<<<(L2N * KP) / TPB, TPB, 0, stream>>>(W4h, W4, L2N, D3, KP);
    cvt_pad<<<(BATCH * KP) / TPB, TPB, 0, stream>>>(datah, data, BATCH, D3, KP);
    // Hoisted loop-invariant GEMM: P2 = data @ W4.T  (bias b4 added in update)
    gemm_wmma_dual<<<dim3(32, 1), TPB, 0, stream>>>(datah, W4h, P2, datah, W4h, P2,
                                                    KP, KP, L2N, L2N, KP);

    // --- 30 Euler steps ---
    for (int t = 0; t < NSTEPS; ++t) {
        spikes_kernel<<<gAll, TPB, 0, stream>>>(s0f, s1f, s2f, sp0h, sp1h, sp2h, rho1h, t);
        // A1 = sp2 @ W2.T  and  A2 = rho(s1) @ W3.T  in one dual launch
        gemm_wmma_dual<<<dim3(32, 2), TPB, 0, stream>>>(sp2h, W2h, A1, rho1h, W3h, A2,
                                                        L1, L1, L1, L1, L1);
        a0_gemm<<<CNT0 / TPB, TPB, 0, stream>>>(sp1h, W0, A0);
        update_kernel<<<gAll, TPB, 0, stream>>>(s0f, s1f, s2f, A0, A1, A2, P2,
                                                b0, b2, b4, sp0h, W1);
    }

    final_copy<<<(CNTOUT + TPB - 1) / TPB, TPB, 0, stream>>>((float*)d_out, s0f, s1f, s2f, data);
}
